// MNIST_SNN_42674795053969
// MI455X (gfx1250) — compile-verified, matrix operand order checked by reading, not run on hardware
//
#include <hip/hip_runtime.h>

// ---------------------------------------------------------------------------
// MNIST SNN (2-layer LIF, snntorch Leaky semantics), CDNA5 / gfx1250.
//   K0: pre-split W1 (fp32 -> bf16 hi/lo) into tiled layout [25][2][112][32]
//   K1: cur1[102400,100] = x[102400,784] @ W1.T + b1
//       split-bf16 WMMA (Ahi*Bhi + Alo*Bhi + Ahi*Blo), f32 accumulate,
//       double-buffered LDS + register prefetch 2 tiles ahead.
//   K2: layer-1 LIF recurrence over t, in-place cur1 -> spk1
//   K3: cur2[102400,16]  = spk1 @ W2.T + b2   (fp32 WMMA 16x16x4, exact)
//   K4: layer-2 LIF recurrence -> out[T,B,10]
// ---------------------------------------------------------------------------

typedef __attribute__((ext_vector_type(16))) __bf16          v16bf;
typedef __attribute__((ext_vector_type(16))) unsigned short  v16us;
typedef __attribute__((ext_vector_type(8)))  float           v8f;
typedef __attribute__((ext_vector_type(2)))  float           v2f;

#define BATCH   1024
#define TSTEPS  100
#define ROWS    (BATCH * TSTEPS)   // 102400 flattened (b,t) rows
#define KDIM1   784
#define NDIM1   100
#define NT1     7                  // 7 N-tiles of 16 (112 >= 100)
#define KT1     25                 // 25 K-tiles of 32 (800 >= 784)
#define NPAD2   16                 // layer-2 N padded 10 -> 16
#define BTILE   3584               // 112*32 elements per (hi or lo) W1 K-tile

// Round-to-nearest-even fp32 -> bf16 split: f ~= hi + lo (both bf16).
__device__ __forceinline__ void bf16_split(float f, unsigned short& hi, unsigned short& lo) {
    unsigned u = __float_as_uint(f);
    unsigned r = u + 0x7FFFu + ((u >> 16) & 1u);
    hi = (unsigned short)(r >> 16);
    float hf = __uint_as_float((unsigned)hi << 16);
    float d  = f - hf;
    unsigned v = __float_as_uint(d);
    unsigned s = v + 0x7FFFu + ((v >> 16) & 1u);
    lo = (unsigned short)(s >> 16);
}

// Load one 16x32 bf16 fragment from an LDS tile (row-major, stride 32 elems)
// into the ISA 16-bit A/B VGPR layout:
//   lanes 0-15 : M = lane,    VGPR v holds K = {0,1},{2,3},{4,5},{6,7},{16,17},...
//   lanes 16-31: M = lane-16, same pattern shifted by +8 in K.
__device__ __forceinline__ v16bf load_frag16x32(const unsigned short* t, int lane) {
    int m = lane & 15, half = lane >> 4;
    v16us u;
#pragma unroll
    for (int v = 0; v < 8; ++v) {
        int k = ((v < 4) ? (2 * v) : (16 + 2 * (v - 4))) + 8 * half;
        u[2 * v]     = t[m * 32 + k];
        u[2 * v + 1] = t[m * 32 + k + 1];
    }
    return __builtin_bit_cast(v16bf, u);
}

// ---------------------------------------------------------------------------
// K0: one-shot W1 split into tiled bf16 hi/lo: w1s[kt][h][n][k], n padded to
// 112 (zeros), K padded to 800 (zeros). 25*2*112*32 ushorts = 350 KB.
// ---------------------------------------------------------------------------
__global__ void snn_split_w1(const float* __restrict__ W1, unsigned short* __restrict__ w1s)
{
    const int i = blockIdx.x * blockDim.x + threadIdx.x;
    if (i >= KT1 * BTILE) return;
    const int kt  = i / BTILE;
    const int rem = i - kt * BTILE;      // n*32 + k
    const int n   = rem >> 5;
    const int k   = rem & 31;
    const int kk  = kt * 32 + k;
    const float f = (n < NDIM1 && kk < KDIM1) ? W1[(size_t)n * KDIM1 + kk] : 0.0f;
    unsigned short h, l;
    bf16_split(f, h, l);
    w1s[(size_t)(kt * 2 + 0) * BTILE + rem] = h;
    w1s[(size_t)(kt * 2 + 1) * BTILE + rem] = l;
}

// ---------------------------------------------------------------------------
// K1: cur1 = x @ W1.T + b1. Block = 256 threads = 8 waves; block owns a
// 128-row M strip, wave owns 16 M-rows x 7 N-tiles. Double-buffered LDS,
// global->register prefetch two K-tiles ahead so HBM traffic overlaps WMMA.
// ---------------------------------------------------------------------------
__global__ __launch_bounds__(256) void snn_gemm1_wmma(
    const float* __restrict__ x, const unsigned short* __restrict__ w1s,
    const float* __restrict__ b1, float* __restrict__ cur1)
{
    __shared__ unsigned short sA[2][2][128 * 32];   // [buf][hi/lo] x-tile
    __shared__ unsigned short sB[2][2 * BTILE];     // [buf] {hi,lo} W1-tile

    const int tid  = threadIdx.x;
    const int wave = tid >> 5;
    const int lane = tid & 31;
    const int row0 = blockIdx.x * 128;
    const int r    = tid >> 1;             // x-tile row owned by this thread
    const int ks   = (tid & 1) * 16;       // 16-wide K chunk within row

    float4 ax[4];       // prefetched x (16 floats)
    uint4  rb[4];       // prefetched pre-split W1 tile slice (896 uint4 / block)

    auto PFX = [&](int kt) {
        const int k0 = kt * 32;
        if (k0 + ks < KDIM1) {
            const float4* s = reinterpret_cast<const float4*>(
                x + (size_t)(row0 + r) * KDIM1 + k0 + ks);
            ax[0] = s[0]; ax[1] = s[1]; ax[2] = s[2]; ax[3] = s[3];
        } else {
            ax[0] = ax[1] = ax[2] = ax[3] = make_float4(0.f, 0.f, 0.f, 0.f);
        }
        const uint4* sb = reinterpret_cast<const uint4*>(w1s + (size_t)kt * 2 * BTILE);
#pragma unroll
        for (int c = 0; c < 4; ++c) {
            const int i = tid + c * 256;
            if (i < 2 * BTILE / 8) rb[c] = sb[i];   // 896 uint4 total
        }
    };
    auto STORE = [&](int p) {
        const float* vf = reinterpret_cast<const float*>(ax);
#pragma unroll
        for (int j = 0; j < 16; ++j) {
            unsigned short h, l;
            bf16_split(vf[j], h, l);
            sA[p][0][r * 32 + ks + j] = h;
            sA[p][1][r * 32 + ks + j] = l;
        }
        uint4* db = reinterpret_cast<uint4*>(sB[p]);
#pragma unroll
        for (int c = 0; c < 4; ++c) {
            const int i = tid + c * 256;
            if (i < 2 * BTILE / 8) db[i] = rb[c];
        }
    };

    v8f acc[NT1] = {};

    PFX(0);
    STORE(0);
    PFX(1);
    __syncthreads();

    for (int kt = 0; kt < KT1; ++kt) {
        const int p = kt & 1;
        if (kt < KT1 - 1) STORE(1 - p);        // stage next tile (other buffer)
        if (kt < KT1 - 2) PFX(kt + 2);         // HBM loads overlap WMMAs below

        const v16bf ahi = load_frag16x32(sA[p][0] + wave * 512, lane);
        const v16bf alo = load_frag16x32(sA[p][1] + wave * 512, lane);
        const unsigned short* bh = sB[p];
        const unsigned short* bl = sB[p] + BTILE;
#pragma unroll
        for (int nt = 0; nt < NT1; ++nt) {
            const v16bf bhi = load_frag16x32(bh + nt * 512, lane);
            const v16bf blo = load_frag16x32(bl + nt * 512, lane);
            acc[nt] = __builtin_amdgcn_wmma_f32_16x16x32_bf16(
                false, ahi, false, bhi, (short)0, acc[nt], false, false);
            acc[nt] = __builtin_amdgcn_wmma_f32_16x16x32_bf16(
                false, alo, false, bhi, (short)0, acc[nt], false, false);
            acc[nt] = __builtin_amdgcn_wmma_f32_16x16x32_bf16(
                false, ahi, false, blo, (short)0, acc[nt], false, false);
        }
        __syncthreads();
    }

    // Epilogue: bias + store. C/D layout: col N = lane%16, row M = v + 8*(lane/16).
    const int cidx = lane & 15;
    const int half = lane >> 4;
    const int growbase = row0 + wave * 16;
#pragma unroll
    for (int nt = 0; nt < NT1; ++nt) {
        const int col = nt * 16 + cidx;
        if (col < NDIM1) {
            const float bias = b1[col];
#pragma unroll
            for (int v = 0; v < 8; ++v) {
                const int row = growbase + v + 8 * half;
                cur1[(size_t)row * NDIM1 + col] = acc[nt][v] + bias;
            }
        }
    }
}

// ---------------------------------------------------------------------------
// K2: layer-1 LIF recurrence. One thread per (batch, neuron) = 102400 threads;
// serial in t (true dependency), coalesced across neuron index. Writes spikes
// in place over cur1 (each thread exclusively owns its addresses).
// ---------------------------------------------------------------------------
__global__ void snn_lif1(float* __restrict__ cur1spk)
{
    const int tid = blockIdx.x * blockDim.x + threadIdx.x;
    if (tid >= BATCH * NDIM1) return;
    const int b = tid / NDIM1;
    const int n = tid - b * NDIM1;
    float mem = 0.0f, spk = 0.0f;
    size_t idx = (size_t)b * TSTEPS * NDIM1 + n;
    for (int t = 0; t < TSTEPS; ++t) {
        const float cur = cur1spk[idx];
        mem = 0.9f * mem + cur - spk;           // beta*mem + I - spk_prev*thresh
        spk = (mem - 1.0f > 0.0f) ? 1.0f : 0.0f;
        cur1spk[idx] = spk;
        idx += NDIM1;
    }
}

// ---------------------------------------------------------------------------
// K3: cur2 = spk1 @ W2.T + b2 via V_WMMA_F32_16X16X4_F32 (exact fp32: spikes
// are {0,1}). One wave per 16x16 output tile, K = 100 = 25 k-tiles of 4.
// fp32 A layout: a[0]=A[m][k0+2*half], a[1]=A[m][k0+2*half+1].
// ---------------------------------------------------------------------------
__global__ __launch_bounds__(256) void snn_gemm2_wmma(
    const float* __restrict__ spk1, const float* __restrict__ W2,
    const float* __restrict__ b2, float* __restrict__ cur2)
{
    __shared__ float sB[NPAD2 * NDIM1];     // W2 zero-padded to 16 rows
    const int tid = threadIdx.x;
    for (int i = tid; i < NPAD2 * NDIM1; i += 256) {
        const int n = i / NDIM1, k = i - n * NDIM1;
        sB[i] = (n < 10) ? W2[n * NDIM1 + k] : 0.0f;
    }
    __syncthreads();

    const int wave = tid >> 5, lane = tid & 31;
    const int m = lane & 15, half = lane >> 4;
    const size_t row0 = ((size_t)blockIdx.x * 8 + wave) * 16;

    const float* arow = spk1 + (row0 + m) * NDIM1;
    const float* brow = sB + m * NDIM1;

    v8f acc = {};
#pragma unroll 5
    for (int kt = 0; kt < 25; ++kt) {
        const int k = kt * 4 + 2 * half;
        v2f a, b;
        a[0] = arow[k];     a[1] = arow[k + 1];
        b[0] = brow[k];     b[1] = brow[k + 1];
        acc = __builtin_amdgcn_wmma_f32_16x16x4_f32(
            false, a, false, b, (short)0, acc, false, false);
    }

    const float bias = (m < 10) ? b2[m] : 0.0f;
#pragma unroll
    for (int v = 0; v < 8; ++v) {
        const size_t row = row0 + v + 8 * half;
        cur2[row * NPAD2 + m] = acc[v] + bias;
    }
}

// ---------------------------------------------------------------------------
// K4: layer-2 LIF recurrence. One thread per (batch, output) = 10240 threads.
// Writes final spikes to out[t, b, i].
// ---------------------------------------------------------------------------
__global__ void snn_lif2(const float* __restrict__ cur2, float* __restrict__ out)
{
    const int tid = blockIdx.x * blockDim.x + threadIdx.x;
    if (tid >= BATCH * 10) return;
    const int b = tid / 10;
    const int i = tid - b * 10;
    float mem = 0.0f, spk = 0.0f;
    for (int t = 0; t < TSTEPS; ++t) {
        const float c = cur2[((size_t)b * TSTEPS + t) * NPAD2 + i];
        mem = 0.9f * mem + c - spk;
        spk = (mem - 1.0f > 0.0f) ? 1.0f : 0.0f;
        out[((size_t)t * BATCH + b) * 10 + i] = spk;
    }
}

// ---------------------------------------------------------------------------
extern "C" void kernel_launch(void* const* d_in, const int* in_sizes, int n_in,
                              void* d_out, int out_size, void* d_ws, size_t ws_size,
                              hipStream_t stream)
{
    (void)in_sizes; (void)n_in; (void)out_size; (void)ws_size;
    const float* x  = (const float*)d_in[0];   // [1024, 100, 784]
    const float* W1 = (const float*)d_in[1];   // [100, 784]
    const float* b1 = (const float*)d_in[2];   // [100]
    const float* W2 = (const float*)d_in[3];   // [10, 100]
    const float* b2 = (const float*)d_in[4];   // [10]
    float* out = (float*)d_out;                // [100, 1024, 10]

    char* ws = (char*)d_ws;
    float* cur1 = (float*)ws;                                       // 102400*100 f32 (~41 MB)
    float* cur2 = (float*)(ws + (size_t)ROWS * NDIM1 * sizeof(float));          // 102400*16 f32
    unsigned short* w1s = (unsigned short*)(ws + (size_t)ROWS * NDIM1 * sizeof(float)
                                               + (size_t)ROWS * NPAD2 * sizeof(float)); // 350 KB

    snn_split_w1<<<(KT1 * BTILE + 255) / 256, 256, 0, stream>>>(W1, w1s);
    snn_gemm1_wmma<<<ROWS / 128, 256, 0, stream>>>(x, w1s, b1, cur1);
    snn_lif1<<<(BATCH * NDIM1 + 255) / 256, 256, 0, stream>>>(cur1);
    snn_gemm2_wmma<<<ROWS / (16 * 8), 256, 0, stream>>>(cur1, W2, b2, cur2);
    snn_lif2<<<(BATCH * 10 + 255) / 256, 256, 0, stream>>>(cur2, out);
}
